// MultiHeadAttention_72232759984804
// MI455X (gfx1250) — matrix operand order repaired and checked by reading, not compiled, not run on hardware
//
#include <hip/hip_runtime.h>

// ---------------------------------------------------------------------------
// MHA for gfx1250: bf16 WMMA (v_wmma_f32_16x16x32_bf16), f32 accumulate.
// B=8, L=1024, D=1024, H=16, hd=64.
// GEMMs: 128x64 block tile, K-step 64, double-buffered LDS (1 barrier/step),
// 8 WMMAs per wave per step.  Attention: flash-style, 16 q-rows per wave.
// f32->bf16 staging uses packed v_cvt_pk_bf16_f32 (one VALU op per dword).
// ---------------------------------------------------------------------------

typedef __attribute__((ext_vector_type(16))) __bf16 v16bf;
typedef __attribute__((ext_vector_type(2)))  __bf16 v2bf;
typedef __attribute__((ext_vector_type(8)))  float  v8f;

#define D_MODEL   1024
#define SEQ_L     1024
#define NUM_HEADS 16
#define HEAD_DIM  64
#define BATCH     8
#define M_TOT     (BATCH * SEQ_L)   // 8192
#define TD        (3 * D_MODEL)     // 3072

// Pack two f32 into one dword of bf16 (lo -> [15:0], hi -> [31:16]).
static __device__ __forceinline__ unsigned pack_bf16(float lo, float hi) {
#if __has_builtin(__builtin_amdgcn_cvt_pk_bf16_f32)
    v2bf v = __builtin_amdgcn_cvt_pk_bf16_f32(lo, hi);
#else
    v2bf v = { (__bf16)lo, (__bf16)hi };
#endif
    union { v2bf v; unsigned u; } c; c.v = v;
    return c.u;
}
static __device__ __forceinline__ unsigned short f32_to_bf16(float f) {
    union { __bf16 h; unsigned short s; } c;
    c.h = (__bf16)f;
    return c.s;
}
static __device__ __forceinline__ uint2 cvt4(float4 f) {
    uint2 u;
    u.x = pack_bf16(f.x, f.y);
    u.y = pack_bf16(f.z, f.w);
    return u;
}

union BF16Frag { v16bf v; uint4 u[2]; };

// Load 16 bf16 as two 16-byte chunks (elements 0-7 from p0, 8-15 from p1).
static __device__ __forceinline__ v16bf load16bf(const unsigned short* p0,
                                                 const unsigned short* p1) {
    BF16Frag r;
    r.u[0] = *reinterpret_cast<const uint4*>(p0);
    r.u[1] = *reinterpret_cast<const uint4*>(p1);
    return r.v;
}

static __device__ __forceinline__ v8f wmma_bf16(v16bf a, v16bf b, v8f c) {
    return __builtin_amdgcn_wmma_f32_16x16x32_bf16(
        /*neg_a=*/false, a, /*neg_b=*/false, b,
        /*c_mod=*/(short)0, c, /*reuse_a=*/false, /*reuse_b=*/false);
}

// A fragment (16x32 bf16): lane row = lane&15, kg = lane>>4,
// elems 0-15 -> K = kg*16 + e (matches B fragment order).
static __device__ __forceinline__ v16bf load_a_frag(const unsigned short* rowp, int kg) {
    return load16bf(rowp + kg * 16, rowp + kg * 16 + 8);
}
// B fragment (32x16 bf16): lane col = lane&15, K = (lane>>4)*16 + 0..15.
static __device__ __forceinline__ v16bf load_b_frag(const unsigned short* colp) {
    return load16bf(colp, colp + 8);
}

// ---------------------------------------------------------------------------
// Kernel 1: fused QKV projection.  C[i,j] = x[i,:] . qkv_w[j,:] + qkv_b[j]
// Scatter to Q/K row-major [bh][l][d] (Q pre-scaled 1/8) and Vt [bh][d][l].
// ---------------------------------------------------------------------------
__global__ __launch_bounds__(256)
void qkv_gemm_kernel(const float* __restrict__ x, const float* __restrict__ w,
                     const float* __restrict__ bias,
                     unsigned short* __restrict__ Q,
                     unsigned short* __restrict__ K,
                     unsigned short* __restrict__ Vt) {
    __shared__ unsigned short As[2][128 * 64];   // 32 KB
    __shared__ unsigned short Bs[2][64 * 64];    // 16 KB

    const int tid  = threadIdx.x;
    const int lane = tid & 31, wv = tid >> 5;
    const int wm = wv & 3, wn = wv >> 2;         // 4 x 2 wave grid, 32x32 tiles
    const int kg = lane >> 4, ln = lane & 15;
    const int bm = blockIdx.x, bn = blockIdx.y;

    const float* xBase = x + bm * 128 * D_MODEL;
    const float* wBase = w + bn * 64 * D_MODEL;

    float4 fa[8], fb[4];                         // raw f32 staging
    auto load_stage = [&](int kk) {
        #pragma unroll
        for (int i = 0; i < 8; ++i) {            // A: 128x64 = 2048 float4
            int t = tid + i * 256;
            int r = t >> 4, c4 = (t & 15) * 4;
            fa[i] = *reinterpret_cast<const float4*>(xBase + r * D_MODEL + kk + c4);
        }
        #pragma unroll
        for (int i = 0; i < 4; ++i) {            // B: 64x64 = 1024 float4
            int t = tid + i * 256;
            int r = t >> 4, c4 = (t & 15) * 4;
            fb[i] = *reinterpret_cast<const float4*>(wBase + r * D_MODEL + kk + c4);
        }
    };
    auto store_stage = [&](int buf) {
        #pragma unroll
        for (int i = 0; i < 8; ++i) {
            int t = tid + i * 256;
            int r = t >> 4, c4 = (t & 15) * 4;
            *reinterpret_cast<uint2*>(&As[buf][r * 64 + c4]) = cvt4(fa[i]);
        }
        #pragma unroll
        for (int i = 0; i < 4; ++i) {
            int t = tid + i * 256;
            int r = t >> 4, c4 = (t & 15) * 4;
            *reinterpret_cast<uint2*>(&Bs[buf][r * 64 + c4]) = cvt4(fb[i]);
        }
    };

    v8f acc[2][2] = {};
    load_stage(0);
    store_stage(0);
    __syncthreads();

    int buf = 0;
    for (int kk = 0; kk < D_MODEL; kk += 64) {
        const bool has_next = (kk + 64) < D_MODEL;
        if (has_next) load_stage(kk + 64);       // overlap with WMMAs below

        #pragma unroll
        for (int kc = 0; kc < 2; ++kc) {         // two 32-wide K chunks
            v16bf af[2], bfr[2];
            #pragma unroll
            for (int tm = 0; tm < 2; ++tm)
                af[tm] = load_a_frag(&As[buf][(wm * 32 + tm * 16 + ln) * 64 + kc * 32], kg);
            #pragma unroll
            for (int tn = 0; tn < 2; ++tn)
                bfr[tn] = load_b_frag(&Bs[buf][(wn * 32 + tn * 16 + ln) * 64 + kc * 32 + kg * 16]);
            #pragma unroll
            for (int tm = 0; tm < 2; ++tm)
                #pragma unroll
                for (int tn = 0; tn < 2; ++tn)
                    acc[tm][tn] = wmma_bf16(af[tm], bfr[tn], acc[tm][tn]);
        }

        if (has_next) {
            store_stage(buf ^ 1);                // write other buffer: safe,
            __syncthreads();                     // readers left it last step, before the barrier
            buf ^= 1;
        }
    }

    // Epilogue: bias, scale Q, scatter.
    #pragma unroll
    for (int tm = 0; tm < 2; ++tm) {
        #pragma unroll
        for (int tn = 0; tn < 2; ++tn) {
            int rowbase = bm * 128 + wm * 32 + tm * 16 + 8 * kg;
            int col     = bn * 64 + wn * 32 + tn * 16 + ln;     // j in [0,3072)
            float bj = bias[col];
            int sel = col >> 10;            // 0=Q 1=K 2=V
            int jj  = col & 1023;
            int h   = jj >> 6, d = jj & 63;
            #pragma unroll
            for (int r = 0; r < 8; ++r) {
                int i = rowbase + r;
                int b = i >> 10, l = i & 1023;
                int bh = b * NUM_HEADS + h;
                float v = acc[tm][tn][r] + bj;
                if (sel == 0) {
                    Q[(bh * SEQ_L + l) * HEAD_DIM + d] = f32_to_bf16(v * 0.125f);
                } else if (sel == 1) {
                    K[(bh * SEQ_L + l) * HEAD_DIM + d] = f32_to_bf16(v);
                } else {
                    Vt[(bh * HEAD_DIM + d) * SEQ_L + l] = f32_to_bf16(v);
                }
            }
        }
    }
}

// ---------------------------------------------------------------------------
// Kernel 2: causal flash attention.  Grid (B*H, L/128); each wave owns 16
// query rows, streams key blocks of 32 with online softmax.
// ---------------------------------------------------------------------------
__global__ __launch_bounds__(256)
void attn_kernel(const unsigned short* __restrict__ Q,
                 const unsigned short* __restrict__ K,
                 const unsigned short* __restrict__ Vt,
                 unsigned short* __restrict__ Ctx) {
    __shared__ unsigned short Ps[8][16 * 32];    // per-wave P staging, 8 KB

    const int bh   = blockIdx.x;                 // b*16 + h
    const int lane = threadIdx.x & 31, wv = threadIdx.x >> 5;
    const int kg = lane >> 4, ln = lane & 15;
    const int q0 = blockIdx.y * 128 + wv * 16;

    const unsigned short* qp = Q  + (size_t)bh * SEQ_L * HEAD_DIM;
    const unsigned short* kp = K  + (size_t)bh * SEQ_L * HEAD_DIM;
    const unsigned short* vp = Vt + (size_t)bh * HEAD_DIM * SEQ_L;

    v8f o[4] = {};
    float mrow[8], lrow[8];
    #pragma unroll
    for (int r = 0; r < 8; ++r) { mrow[r] = -3.0e38f; lrow[r] = 0.0f; }

    const unsigned short* qrowp = qp + (q0 + ln) * HEAD_DIM;
    const int kblocks = (q0 + 15) / 32 + 1;

    for (int kb = 0; kb < kblocks; ++kb) {
        // S = Q Kt : two 16x16 tiles covering keys [kb*32, kb*32+32)
        v8f s0 = {}, s1 = {};
        #pragma unroll
        for (int ch = 0; ch < HEAD_DIM; ch += 32) {
            v16bf aq = load_a_frag(qrowp + ch, kg);
            const unsigned short* k0p = kp + (kb * 32 + ln) * HEAD_DIM + ch + kg * 16;
            const unsigned short* k1p = k0p + 16 * HEAD_DIM;
            s0 = wmma_bf16(aq, load_b_frag(k0p), s0);
            s1 = wmma_bf16(aq, load_b_frag(k1p), s1);
        }

        // Online softmax update (rows r+8*kg live in this lane's half-wave).
        const int key0 = kb * 32 + ln, key1 = key0 + 16;
        float alpha[8];
        #pragma unroll
        for (int r = 0; r < 8; ++r) {
            int qrow = q0 + r + 8 * kg;
            float v0 = (key0 <= qrow) ? s0[r] : -3.0e38f;
            float v1 = (key1 <= qrow) ? s1[r] : -3.0e38f;
            float mx = fmaxf(v0, v1);
            #pragma unroll
            for (int off = 1; off < 16; off <<= 1)
                mx = fmaxf(mx, __shfl_xor(mx, off, 32));
            float mn = fmaxf(mrow[r], mx);
            float a  = __expf(mrow[r] - mn);
            float p0 = __expf(v0 - mn);
            float p1 = __expf(v1 - mn);
            float ps = p0 + p1;
            #pragma unroll
            for (int off = 1; off < 16; off <<= 1)
                ps += __shfl_xor(ps, off, 32);
            lrow[r]  = lrow[r] * a + ps;
            mrow[r]  = mn;
            alpha[r] = a;
            unsigned short* pr = &Ps[wv][(r + 8 * kg) * 32];
            pr[ln]      = f32_to_bf16(p0);
            pr[16 + ln] = f32_to_bf16(p1);
        }
        #pragma unroll
        for (int c = 0; c < 4; ++c)
            #pragma unroll
            for (int r = 0; r < 8; ++r)
                o[c][r] *= alpha[r];

        // O += P V  (P from per-wave LDS in A layout; V via Vt, K-contiguous)
        v16bf pf = load_a_frag(&Ps[wv][ln * 32], kg);
        #pragma unroll
        for (int c = 0; c < 4; ++c) {
            const unsigned short* vcol =
                vp + (c * 16 + ln) * SEQ_L + kb * 32 + kg * 16;
            o[c] = wmma_bf16(pf, load_b_frag(vcol), o[c]);
        }
    }

    // Normalize and store ctx[b, l, h*64 + d] as bf16.
    const int b = bh >> 4, h = bh & 15;
    #pragma unroll
    for (int c = 0; c < 4; ++c) {
        #pragma unroll
        for (int r = 0; r < 8; ++r) {
            int qrow = q0 + r + 8 * kg;
            int col  = h * HEAD_DIM + c * 16 + ln;
            float v  = o[c][r] / lrow[r];
            Ctx[(b * SEQ_L + qrow) * D_MODEL + col] = f32_to_bf16(v);
        }
    }
}

// ---------------------------------------------------------------------------
// Kernel 3: output projection.  out[i,j] = ctx[i,:] . out_w[j,:] + out_b[j]
// Same double-buffered scheme; A (ctx) is already bf16.
// ---------------------------------------------------------------------------
__global__ __launch_bounds__(256)
void out_gemm_kernel(const unsigned short* __restrict__ ctx,
                     const float* __restrict__ w, const float* __restrict__ bias,
                     float* __restrict__ out) {
    __shared__ unsigned short As[2][128 * 64];   // 32 KB
    __shared__ unsigned short Bs[2][64 * 64];    // 16 KB

    const int tid  = threadIdx.x;
    const int lane = tid & 31, wv = tid >> 5;
    const int wm = wv & 3, wn = wv >> 2;
    const int kg = lane >> 4, ln = lane & 15;
    const int bm = blockIdx.x, bn = blockIdx.y;

    const unsigned short* aBase = ctx + bm * 128 * D_MODEL;
    const float*          wBase = w   + bn * 64 * D_MODEL;

    uint4  ua[2];                                // A: 128x32 bf16 = 512 uint4
    float4 fb[4];                                // B: 64x64 f32  = 1024 float4
    auto load_stage = [&](int kk) {
        #pragma unroll
        for (int i = 0; i < 2; ++i) {
            int t = tid + i * 256;
            int r = t >> 2, c8 = (t & 3) * 8;
            ua[i] = *reinterpret_cast<const uint4*>(aBase + r * D_MODEL + kk + c8);
        }
        #pragma unroll
        for (int i = 0; i < 4; ++i) {
            int t = tid + i * 256;
            int r = t >> 4, c4 = (t & 15) * 4;
            fb[i] = *reinterpret_cast<const float4*>(wBase + r * D_MODEL + kk + c4);
        }
    };
    auto store_stage = [&](int buf) {
        #pragma unroll
        for (int i = 0; i < 2; ++i) {
            int t = tid + i * 256;
            int r = t >> 2, c8 = (t & 3) * 8;
            *reinterpret_cast<uint4*>(&As[buf][r * 64 + c8]) = ua[i];
        }
        #pragma unroll
        for (int i = 0; i < 4; ++i) {
            int t = tid + i * 256;
            int r = t >> 4, c4 = (t & 15) * 4;
            *reinterpret_cast<uint2*>(&Bs[buf][r * 64 + c4]) = cvt4(fb[i]);
        }
    };

    v8f acc[2][2] = {};
    load_stage(0);
    store_stage(0);
    __syncthreads();

    int buf = 0;
    for (int kk = 0; kk < D_MODEL; kk += 64) {
        const bool has_next = (kk + 64) < D_MODEL;
        if (has_next) load_stage(kk + 64);

        #pragma unroll
        for (int kc = 0; kc < 2; ++kc) {
            v16bf af[2], bfr[2];
            #pragma unroll
            for (int tm = 0; tm < 2; ++tm)
                af[tm] = load_a_frag(&As[buf][(wm * 32 + tm * 16 + ln) * 64 + kc * 32], kg);
            #pragma unroll
            for (int tn = 0; tn < 2; ++tn)
                bfr[tn] = load_b_frag(&Bs[buf][(wn * 32 + tn * 16 + ln) * 64 + kc * 32 + kg * 16]);
            #pragma unroll
            for (int tm = 0; tm < 2; ++tm)
                #pragma unroll
                for (int tn = 0; tn < 2; ++tn)
                    acc[tm][tn] = wmma_bf16(af[tm], bfr[tn], acc[tm][tn]);
        }

        if (has_next) {
            store_stage(buf ^ 1);
            __syncthreads();
            buf ^= 1;
        }
    }

    #pragma unroll
    for (int tm = 0; tm < 2; ++tm) {
        #pragma unroll
        for (int tn = 0; tn < 2; ++tn) {
            int rowbase = bm * 128 + wm * 32 + tm * 16 + 8 * kg;
            int col     = bn * 64 + wn * 32 + tn * 16 + ln;
            float bj = bias[col];
            #pragma unroll
            for (int r = 0; r < 8; ++r)
                out[(rowbase + r) * D_MODEL + col] = acc[tm][tn][r] + bj;
        }
    }
}

// ---------------------------------------------------------------------------
extern "C" void kernel_launch(void* const* d_in, const int* in_sizes, int n_in,
                              void* d_out, int out_size, void* d_ws, size_t ws_size,
                              hipStream_t stream) {
    const float* x      = (const float*)d_in[0];
    // d_in[1] = causal mask (handled analytically)
    const float* qkv_w  = (const float*)d_in[2];
    const float* qkv_b  = (const float*)d_in[3];
    const float* out_w  = (const float*)d_in[4];
    const float* out_b  = (const float*)d_in[5];
    float*       out    = (float*)d_out;

    const size_t n_qkv = (size_t)BATCH * NUM_HEADS * SEQ_L * HEAD_DIM;  // 8.4M
    unsigned short* Q   = (unsigned short*)d_ws;
    unsigned short* K   = Q  + n_qkv;
    unsigned short* Vt  = K  + n_qkv;
    unsigned short* Ctx = Vt + n_qkv;   // [8192,1024] bf16

    qkv_gemm_kernel<<<dim3(M_TOT / 128, TD / 64), 256, 0, stream>>>(
        x, qkv_w, qkv_b, Q, K, Vt);
    attn_kernel<<<dim3(BATCH * NUM_HEADS, SEQ_L / 128), 256, 0, stream>>>(
        Q, K, Vt, Ctx);
    out_gemm_kernel<<<dim3(M_TOT / 128, D_MODEL / 64), 256, 0, stream>>>(
        Ctx, out_w, out_b, out);
}
